// CategoryBranch2_33165737459881
// MI455X (gfx1250) — matrix operand order repaired and checked by reading, not compiled, not run on hardware
//
#include <hip/hip_runtime.h>
#include <hip/hip_bf16.h>
#include <math.h>

// ---------------------------------------------------------------------------
// CDNA5 / gfx1250: all GEMMs via v_wmma_f32_16x16x32_bf16 (f32 accum).
// Tiles staged in fragment-swizzled LDS layouts so every WMMA fragment is a
// 32-byte contiguous per-lane load (2x b128). Dense-GEMM weight tiles are
// staged with global_load_async_to_lds_b128 (ASYNCcnt) to bypass VGPRs.
// ---------------------------------------------------------------------------

typedef __bf16 bf16_t;
typedef __attribute__((ext_vector_type(16))) __bf16 v16bf;
typedef __attribute__((ext_vector_type(8)))  __bf16 v8bf;
typedef __attribute__((ext_vector_type(8)))  float  v8f;

#define DEV __device__ __forceinline__

DEV v8f wmma_bf16f32(v16bf a, v16bf b, v8f c) {
  // 8 args: (neg_a, A, neg_b, B, c_mod, C, reuse_a, reuse_b)
  return __builtin_amdgcn_wmma_f32_16x16x32_bf16(false, a, false, b, (short)0, c,
                                                 false, false);
}

// A fragment order used here (16-bit A 16x32, contiguous k; B uses the same):
//   lane<16 : M=lane,    K = {0..15}
//   lane>=16: M=lane-16, K = {16..31}
// We store A tiles pre-swizzled: element (m,k) lives at
//   idx = (((mt*KC + kc)*2 + lh)*16 + row)*16 + e
// with mt=m>>4,row=m&15,kc=k>>5,kk=k&31,lh=kk>>4,e=kk&15,
// so a lane's 16 fragment values are 32 contiguous bytes.
DEV int a_sw_idx(int KC, int m, int k) {
  int mt = m >> 4, row = m & 15;
  int kc = k >> 5, kk = k & 31;
  int lh = kk >> 4;
  int e  = kk & 15;
  return (((mt * KC + kc) * 2 + lh) * 16 + row) * 16 + e;
}

DEV v16bf load_A_frag_sw(const bf16_t* A_sw, int KC, int mt, int kc) {
  const int lane = threadIdx.x & 31;
  const int lh = lane >> 4, row = lane & 15;
  return *(const v16bf*)&A_sw[(((mt * KC + kc) * 2 + lh) * 16 + row) * 16];
}

// B fragment (32x16): b[e] = B[kb+e][col], kb = (lane>=16)*16, col = lane&15.
// From an LDS tile stored n-major Bt[64][32] (Bt[n][k] = Wt[Nbase+n][kc*32+k]):
// each lane reads 16 contiguous bf16 (32B).
DEV v16bf load_B_frag_t(const bf16_t* Bt, int n0) {
  const int lane = threadIdx.x & 31;
  const int col = lane & 15;
  const int kb = (lane >> 4) << 4;
  return *(const v16bf*)&Bt[(n0 + col) * 32 + kb];
}

// B fragment straight from a row-major bf16 weight Wt[N][K]: contiguous in K.
DEV v16bf load_B_frag_wt(const bf16_t* wt, int ldk, int nbase, int kbase) {
  const int lane = threadIdx.x & 31;
  const int col = lane & 15;
  const int kb = kbase + ((lane >> 4) << 4);
  return *(const v16bf*)&wt[(long)(nbase + col) * ldk + kb];
}

// gfx1250 async global -> LDS copy (16B per lane), tracked by ASYNCcnt.
DEV void async_copy_b128(unsigned lds_off, const bf16_t* gaddr) {
  asm volatile("global_load_async_to_lds_b128 %0, %1, off"
               :: "v"(lds_off), "v"(gaddr) : "memory");
}
DEV void wait_async0() { asm volatile("s_wait_asynccnt 0" ::: "memory"); }

DEV float sigmoidf_(float x) { return 1.0f / (1.0f + __expf(-x)); }

// ---------------------------------------------------------------------------
__global__ __launch_bounds__(256) void k_f32_to_bf16(const float* __restrict__ s,
                                                     bf16_t* __restrict__ d, long n) {
  long i = (long)blockIdx.x * blockDim.x + threadIdx.x;
  if (i < n) d[i] = (bf16_t)s[i];
}

// ---------------------------------------------------------------------------
// Implicit-im2col conv GEMM + bias + ReLU. in: NHWC; wt: (Cout, Cin*9) f32.
// Block tile 64x64, 8 waves (4 M-tiles x 2 N-pairs).
// ---------------------------------------------------------------------------
__global__ __launch_bounds__(256) void k_conv_wmma(const float* __restrict__ in,
                                                   const float* __restrict__ wt,
                                                   const float* __restrict__ bias,
                                                   float* __restrict__ out,
                                                   int H, int W, int Cin, int Cout) {
  const int K  = Cin * 9;
  const int KC = (K + 31) >> 5;
  __shared__ __align__(32) bf16_t As[64 * 32];  // swizzled (KC=1 per stage)
  __shared__ __align__(32) bf16_t Bt[64 * 32];  // n-major transposed tile
  const int tid   = threadIdx.x;
  const int lane  = tid & 31;
  const int wid   = tid >> 5;
  const int Mbase = blockIdx.x * 64;
  const int Nbase = blockIdx.y * 64;
  const int mwave = wid & 3;
  const int nwave = wid >> 2;
  v8f acc0 = {}, acc1 = {};
  for (int kc = 0; kc < KC; ++kc) {
    __syncthreads();
#pragma unroll
    for (int r = 0; r < 8; ++r) {   // stage A: implicit im2col, zero pad
      int idx = r * 256 + tid;
      int row = idx >> 5, col = idx & 31;
      int k = kc * 32 + col;
      float v = 0.0f;
      if (k < K) {
        int m = Mbase + row;
        int w = m % W, h = (m / W) % H, n = m / (W * H);
        int ci = k / 9, rem = k % 9;
        int hin = h + rem / 3 - 1, win = w + rem % 3 - 1;
        if (hin >= 0 && hin < H && win >= 0 && win < W)
          v = in[(((long)n * H + hin) * W + win) * Cin + ci];
      }
      As[a_sw_idx(1, row, col)] = (bf16_t)v;
    }
#pragma unroll
    for (int r = 0; r < 8; ++r) {   // stage B transposed: Bt[n][k]
      int idx = r * 256 + tid;
      int n = idx >> 5, kk = idx & 31;
      int k = kc * 32 + kk;
      float v = (k < K) ? wt[(long)(Nbase + n) * K + k] : 0.0f;
      Bt[n * 32 + kk] = (bf16_t)v;
    }
    __syncthreads();
    v16bf a  = load_A_frag_sw(As, 1, mwave, 0);
    v16bf b0 = load_B_frag_t(Bt, nwave * 32);
    v16bf b1 = load_B_frag_t(Bt, nwave * 32 + 16);
    acc0 = wmma_bf16f32(a, b0, acc0);
    acc1 = wmma_bf16f32(a, b1, acc1);
  }
  const int ro = (lane >> 4) * 8, col = lane & 15;
#pragma unroll
  for (int j = 0; j < 8; ++j) {
    int gm = Mbase + mwave * 16 + ro + j;
    int n0 = Nbase + nwave * 32 + col;
    int n1 = n0 + 16;
    out[(long)gm * Cout + n0] = fmaxf(acc0[j] + bias[n0], 0.0f);
    out[(long)gm * Cout + n1] = fmaxf(acc1[j] + bias[n1], 0.0f);
  }
}

// ---------------------------------------------------------------------------
// Dense GEMM: C[M,N] = A[M,K](f32) @ Wt[N,K](bf16)^T + bias.
// K must be a multiple of 32 (true for all call sites: 2048, 512).
// B tile staged via async global->LDS b128; A staged via float4 loads + cvt.
// ---------------------------------------------------------------------------
__global__ __launch_bounds__(256) void k_gemm_wmma(const float* __restrict__ A,
                                                   const bf16_t* __restrict__ wt,
                                                   const float* __restrict__ bias,
                                                   float* __restrict__ C,
                                                   int M, int N, int K) {
  const int KC = K >> 5;
  __shared__ __align__(32) bf16_t As[64 * 32];  // swizzled
  __shared__ __align__(32) bf16_t Bt[64 * 32];  // n-major transposed
  const int tid = threadIdx.x, lane = tid & 31, wid = tid >> 5;
  const int Mbase = blockIdx.x * 64, Nbase = blockIdx.y * 64;
  const int mwave = wid & 3, nwave = wid >> 2;
  const unsigned bt_base = (unsigned)(uintptr_t)(const void*)Bt;
  // async source/dest for this thread: one 16B segment per k-chunk
  const int bn = tid >> 2, bseg = tid & 3;
  // A staging: one 8-wide k segment per thread
  const int arow = tid >> 2;
  const int ak0 = (tid & 3) * 8;
  v8f acc0 = {}, acc1 = {};
  for (int kc = 0; kc < KC; ++kc) {
    __syncthreads();
    // ---- B tile: async copy, bypasses VGPRs (ASYNCcnt) ----
    async_copy_b128(bt_base + (unsigned)(bn * 32 + bseg * 8) * 2,
                    wt + (long)(Nbase + bn) * K + kc * 32 + bseg * 8);
    // ---- A tile: float4 loads, convert, packed swizzled store ----
    {
      const float4* src =
          (const float4*)&A[(long)(Mbase + arow) * K + kc * 32 + ak0];
      float4 p = src[0], q = src[1];
      v8bf pk;
      pk[0] = (bf16_t)p.x; pk[1] = (bf16_t)p.y;
      pk[2] = (bf16_t)p.z; pk[3] = (bf16_t)p.w;
      pk[4] = (bf16_t)q.x; pk[5] = (bf16_t)q.y;
      pk[6] = (bf16_t)q.z; pk[7] = (bf16_t)q.w;
      *(v8bf*)&As[a_sw_idx(1, arow, ak0)] = pk;   // 8-k segment is contiguous
    }
    if (kc + 1 < KC)  // global_prefetch_b8 for the next A chunk
      __builtin_prefetch(&A[(long)(Mbase + (tid & 63)) * K + (kc + 1) * 32], 0, 1);
    wait_async0();        // own wave's async copies done; barrier covers rest
    __syncthreads();
    v16bf a  = load_A_frag_sw(As, 1, mwave, 0);
    v16bf b0 = load_B_frag_t(Bt, nwave * 32);
    v16bf b1 = load_B_frag_t(Bt, nwave * 32 + 16);
    acc0 = wmma_bf16f32(a, b0, acc0);
    acc1 = wmma_bf16f32(a, b1, acc1);
  }
  const int ro = (lane >> 4) * 8, col = lane & 15;
#pragma unroll
  for (int j = 0; j < 8; ++j) {
    int gm = Mbase + mwave * 16 + ro + j;
    int n0 = Nbase + nwave * 32 + col;
    int n1 = n0 + 16;
    C[(long)gm * N + n0] = acc0[j] + bias[n0];
    C[(long)gm * N + n1] = acc1[j] + bias[n1];
  }
}

// ---------------------------------------------------------------------------
// Per-channel mean / rstd over [M,C]; deterministic fixed-order reduction.
// ---------------------------------------------------------------------------
__global__ __launch_bounds__(256) void k_channel_stats(const float* __restrict__ x,
                                                       float* __restrict__ mean,
                                                       float* __restrict__ rstd,
                                                       long M, int C) {
  __shared__ float s1[256], s2[256];
  const int c = blockIdx.x, tid = threadIdx.x;
  float a = 0.0f, b = 0.0f;
  for (long m = tid; m < M; m += 256) {
    float v = x[m * C + c];
    a += v; b += v * v;
  }
  s1[tid] = a; s2[tid] = b;
  __syncthreads();
  for (int s = 128; s > 0; s >>= 1) {
    if (tid < s) { s1[tid] += s1[tid + s]; s2[tid] += s2[tid + s]; }
    __syncthreads();
  }
  if (tid == 0) {
    float mu = s1[0] / (float)M;
    float var = s2[0] / (float)M - mu * mu;
    mean[c] = mu;
    rstd[c] = rsqrtf(var + 1e-5f);
  }
}

__global__ __launch_bounds__(256) void k_norm_pool(const float* __restrict__ x,
                                                   const float* __restrict__ mean,
                                                   const float* __restrict__ rstd,
                                                   float* __restrict__ y,
                                                   int C, int W2, long total) {
  for (long i = (long)blockIdx.x * blockDim.x + threadIdx.x; i < total;
       i += (long)gridDim.x * blockDim.x) {
    int  c  = (int)(i % C);
    long p  = i / C;
    long w2 = p % W2;
    long m0 = (p / W2) * (2 * W2) + 2 * w2;
    float mu = mean[c], rs = rstd[c];
    float v0 = (x[m0 * C + c] - mu) * rs;
    float v1 = (x[(m0 + 1) * C + c] - mu) * rs;
    y[p * C + c] = fmaxf(v0, v1);
  }
}

__global__ __launch_bounds__(256) void k_transpose_x0(const float* __restrict__ pool,
                                                      float* __restrict__ x0,
                                                      long total) {
  for (long i = (long)blockIdx.x * blockDim.x + threadIdx.x; i < total;
       i += (long)gridDim.x * blockDim.x) {
    int  d = (int)(i & 2047);
    long m = i >> 11;
    int  c = d >> 3, f = d & 7;
    x0[i] = pool[(m * 8 + f) * 256 + c];
  }
}

__global__ __launch_bounds__(256) void k_concat_sub(const float* __restrict__ Hf,
                                                    const float* __restrict__ Hb,
                                                    float* __restrict__ X,
                                                    int Tin, int Tout, int stride,
                                                    long total) {
  for (long i = (long)blockIdx.x * blockDim.x + threadIdx.x; i < total;
       i += (long)gridDim.x * blockDim.x) {
    int  d = (int)(i % 512);
    long t = (i / 512) % Tout;
    long b = i / (512 * (long)Tout);
    long tf = t * stride;
    float v = (d < 256) ? Hf[(b * Tin + tf) * 256 + d]
                        : Hb[(b * Tin + tf) * 256 + (d - 256)];
    X[(b * Tout + t) * 512 + d] = v;
  }
}

// ---------------------------------------------------------------------------
// Persistent bidirectional GRU scan. grid=2 (fwd/bwd), 1024 threads = 32 waves.
// Hidden state LDS-resident in fragment-swizzled bf16 (+ f32 master copy).
// Per step gh = h(32x256) @ whh^T(256x768): 96 WMMA tiles, 3 per wave.
// Reference applies an extra tanh on the GRU cell output.
// ---------------------------------------------------------------------------
__global__ __launch_bounds__(1024) void k_gru_scan(const float* __restrict__ GxF,
                                                   const float* __restrict__ GxB,
                                                   const bf16_t* __restrict__ whhF,
                                                   const bf16_t* __restrict__ whhB,
                                                   const float* __restrict__ bhhF,
                                                   const float* __restrict__ bhhB,
                                                   float* __restrict__ HoutF,
                                                   float* __restrict__ HoutB,
                                                   float* __restrict__ ghws,
                                                   int T) {
  const int dir = blockIdx.x;
  const float*  Gx   = dir ? GxB  : GxF;
  const bf16_t* whh  = dir ? whhB : whhF;
  const float*  bhh  = dir ? bhhB : bhhF;
  float*        Hout = dir ? HoutB : HoutF;
  float*        gh   = ghws + (long)dir * 32 * 768;

  __shared__ float  hF[32 * 256];
  __shared__ __align__(32) bf16_t hB[32 * 256];  // swizzled, KC=8
  const int tid = threadIdx.x, lane = tid & 31, wv = tid >> 5;
  for (int i = tid; i < 32 * 256; i += 1024) { hF[i] = 0.0f; hB[i] = (bf16_t)0.0f; }
  __syncthreads();

  for (int tp = 0; tp < T; ++tp) {
    const int tg = dir ? (T - 1 - tp) : tp;
#pragma unroll
    for (int i = 0; i < 3; ++i) {
      int tile = i * 32 + wv;                 // 2 M-tiles x 48 N-tiles
      int mt = tile & 1, nt = tile >> 1;
      v8f acc = {};
#pragma unroll
      for (int kc = 0; kc < 8; ++kc) {
        v16bf a = load_A_frag_sw(hB, 8, mt, kc);
        v16bf b = load_B_frag_wt(whh, 256, nt * 16, kc * 32);
        acc = wmma_bf16f32(a, b, acc);
      }
      int ro = (lane >> 4) * 8, col = lane & 15;
#pragma unroll
      for (int j = 0; j < 8; ++j)
        gh[(mt * 16 + ro + j) * 768 + nt * 16 + col] = acc[j];
    }
    __threadfence_block();
    __syncthreads();
    {
      const int b = tid >> 5;
      const int j0 = lane * 8;
      const float* gx = Gx + ((long)b * T + tg) * 768;
#pragma unroll
      for (int q = 0; q < 8; ++q) {
        int j = j0 + q;
        float ghr = gh[b * 768 + j] + bhh[j];
        float ghz = gh[b * 768 + 256 + j] + bhh[256 + j];
        float ghn = gh[b * 768 + 512 + j] + bhh[512 + j];
        float r = sigmoidf_(gx[j] + ghr);
        float z = sigmoidf_(gx[256 + j] + ghz);
        float n = tanhf(gx[512 + j] + r * ghn);
        float h = hF[b * 256 + j];
        float hn = tanhf((1.0f - z) * n + z * h);
        hF[b * 256 + j] = hn;
        hB[a_sw_idx(8, b, j)] = (bf16_t)hn;   // 8-elem runs stay contiguous
        Hout[((long)b * T + tp) * 256 + j] = hn;
      }
    }
    __threadfence_block();
    __syncthreads();
  }
}

// ---------------------------------------------------------------------------
// Attention decoder: 16 sequential steps, single block of 1024 threads.
// ---------------------------------------------------------------------------
__global__ __launch_bounds__(1024) void k_decoder(const float* __restrict__ X2,
                                                  const float* __restrict__ att_w,
                                                  const float* __restrict__ att_b,
                                                  const bf16_t* __restrict__ wih,
                                                  const float* __restrict__ bih,
                                                  const bf16_t* __restrict__ whh,
                                                  const float* __restrict__ bhh,
                                                  const float* __restrict__ out_w,
                                                  const float* __restrict__ out_b,
                                                  float* __restrict__ logits_out,
                                                  float* __restrict__ w_out,
                                                  bf16_t* __restrict__ att_bf,
                                                  float* __restrict__ Gbuf,
                                                  float* __restrict__ HNbuf,
                                                  float* __restrict__ wbuf) {
  __shared__ float  hF[32 * 256];
  __shared__ __align__(32) bf16_t hB[32 * 256];  // swizzled, KC=8
  __shared__ float  kap[32], alp[32], bet[32];
  const int tid = threadIdx.x, lane = tid & 31, wv = tid >> 5;
  for (int i = tid; i < 32 * 256; i += 1024) { hF[i] = 0.0f; hB[i] = (bf16_t)0.0f; }
  if (tid < 32) kap[tid] = 0.0f;
  __syncthreads();

  for (int s = 0; s < 16; ++s) {
    // a) ph = exp(hidden @ att_w^T + att_b); kappa += kstep
    if (tid < 96) {
      int b = tid / 3, j = tid % 3;
      float d = att_b[j];
      for (int k = 0; k < 256; ++k) d += att_w[j * 256 + k] * hF[b * 256 + k];
      float e = __expf(d);
      if (j == 0) alp[b] = e; else if (j == 1) bet[b] = e; else kap[b] += e;
    }
    __syncthreads();
    // b) attention window
    for (int i = tid; i < 32 * 128; i += 1024) {
      int b = i >> 7, u = i & 127;
      float du = kap[b] - (float)u;
      float w = alp[b] * __expf(-bet[b] * du * du);
      wbuf[i] = w;
      w_out[((long)b * 16 + s) * 128 + u] = w;
    }
    __threadfence_block();
    __syncthreads();
    // c) attended[b,d] = sum_u w[b,u] * X2[b,u,d] -> swizzled bf16 A (KC=16)
    for (int i = tid; i < 32 * 512; i += 1024) {
      int b = i >> 9, d2 = i & 511;
      float a = 0.0f;
      for (int u = 0; u < 128; ++u)
        a += wbuf[b * 128 + u] * X2[((long)b * 128 + u) * 512 + d2];
      att_bf[a_sw_idx(16, b, d2)] = (bf16_t)a;
    }
    __threadfence_block();
    __syncthreads();
    // d) G = attended @ wih^T + bih  (M=32,N=768,K=512)
#pragma unroll
    for (int i2 = 0; i2 < 3; ++i2) {
      int tile = i2 * 32 + wv;
      int mt = tile & 1, nt = tile >> 1;
      v8f acc = {};
#pragma unroll
      for (int kc = 0; kc < 16; ++kc) {
        v16bf a = load_A_frag_sw(att_bf, 16, mt, kc);
        v16bf b = load_B_frag_wt(wih, 512, nt * 16, kc * 32);
        acc = wmma_bf16f32(a, b, acc);
      }
      int ro = (lane >> 4) * 8, col = lane & 15;
#pragma unroll
      for (int j = 0; j < 8; ++j) {
        int gn = nt * 16 + col;
        Gbuf[(mt * 16 + ro + j) * 768 + gn] = acc[j] + bih[gn];
      }
    }
    __threadfence_block();
    __syncthreads();
    // e) gh = hidden @ whh^T + bhh; fold r/z into G, keep hn separate
#pragma unroll
    for (int i2 = 0; i2 < 3; ++i2) {
      int tile = i2 * 32 + wv;
      int mt = tile & 1, nt = tile >> 1;
      v8f acc = {};
#pragma unroll
      for (int kc = 0; kc < 8; ++kc) {
        v16bf a = load_A_frag_sw(hB, 8, mt, kc);
        v16bf b = load_B_frag_wt(whh, 256, nt * 16, kc * 32);
        acc = wmma_bf16f32(a, b, acc);
      }
      int ro = (lane >> 4) * 8, col = lane & 15;
#pragma unroll
      for (int j = 0; j < 8; ++j) {
        int gm = mt * 16 + ro + j;
        int gn = nt * 16 + col;
        float v = acc[j] + bhh[gn];
        if (gn < 512) Gbuf[gm * 768 + gn] += v;
        else          HNbuf[gm * 256 + (gn - 512)] = v;
      }
    }
    __threadfence_block();
    __syncthreads();
    // f) GRU cell (no extra tanh in decoder)
    {
      int b = tid >> 5, j0 = lane * 8;
#pragma unroll
      for (int q = 0; q < 8; ++q) {
        int j = j0 + q;
        float r = sigmoidf_(Gbuf[b * 768 + j]);
        float z = sigmoidf_(Gbuf[b * 768 + 256 + j]);
        float n = tanhf(Gbuf[b * 768 + 512 + j] + r * HNbuf[b * 256 + j]);
        float h = hF[b * 256 + j];
        float hn = (1.0f - z) * n + z * h;
        hF[b * 256 + j] = hn;
        hB[a_sw_idx(8, b, j)] = (bf16_t)hn;
      }
    }
    __syncthreads();
    // g) logits
    if (tid < 512) {
      int b = tid >> 4, c = tid & 15;
      float d = out_b[c];
      for (int k = 0; k < 256; ++k) d += out_w[c * 256 + k] * hF[b * 256 + k];
      logits_out[((long)b * 16 + s) * 16 + c] = d;
    }
    __syncthreads();
  }
}

// ---------------------------------------------------------------------------
// Host orchestration. Input flattening assumed = setup_inputs() insertion order:
//  0 x | 1 cw0 2 cb0 3 cw1 4 cb1 5 cw2 6 cb2
//  7..10 gru_f0{wih,whh,bih,bhh} | 11..14 gru_f1 | 15..18 gru_b0 | 19..22 gru_b1
// 23 att_w 24 att_b | 25..28 dec{wih,whh,bih,bhh} | 29 out_w 30 out_b | 31 output_len
// ---------------------------------------------------------------------------
extern "C" void kernel_launch(void* const* d_in, const int* in_sizes, int n_in,
                              void* d_out, int out_size, void* d_ws, size_t ws_size,
                              hipStream_t stream) {
  (void)in_sizes; (void)n_in; (void)out_size; (void)ws_size;
  const float* x     = (const float*)d_in[0];
  const float* cw[3] = {(const float*)d_in[1], (const float*)d_in[3], (const float*)d_in[5]};
  const float* cb[3] = {(const float*)d_in[2], (const float*)d_in[4], (const float*)d_in[6]};
  const float* f0_wih = (const float*)d_in[7];
  const float* f0_whh = (const float*)d_in[8];
  const float* f0_bih = (const float*)d_in[9];
  const float* f0_bhh = (const float*)d_in[10];
  const float* f1_wih = (const float*)d_in[11];
  const float* f1_whh = (const float*)d_in[12];
  const float* f1_bih = (const float*)d_in[13];
  const float* f1_bhh = (const float*)d_in[14];
  const float* b0_wih = (const float*)d_in[15];
  const float* b0_whh = (const float*)d_in[16];
  const float* b0_bih = (const float*)d_in[17];
  const float* b0_bhh = (const float*)d_in[18];
  const float* b1_wih = (const float*)d_in[19];
  const float* b1_whh = (const float*)d_in[20];
  const float* b1_bih = (const float*)d_in[21];
  const float* b1_bhh = (const float*)d_in[22];
  const float* att_w  = (const float*)d_in[23];
  const float* att_b  = (const float*)d_in[24];
  const float* dc_wih = (const float*)d_in[25];
  const float* dc_whh = (const float*)d_in[26];
  const float* dc_bih = (const float*)d_in[27];
  const float* dc_bhh = (const float*)d_in[28];
  const float* out_w  = (const float*)d_in[29];
  const float* out_b  = (const float*)d_in[30];

  char* base = (char*)d_ws;
  size_t off = 0;
  auto alloc = [&](size_t bytes) -> char* {
    char* p = base + off;
    off = (off + bytes + 255) & ~(size_t)255;
    return p;
  };
  float* conv_buf = (float*)alloc((size_t)33554432 * 4);
  float* poolA    = (float*)alloc((size_t)16777216 * 4);
  float* poolB    = (float*)alloc((size_t)16777216 * 4);
  float* X0       = (float*)alloc((size_t)16777216 * 4);
  float* GxF0     = (float*)alloc((size_t)6291456 * 4);
  float* GxB0     = (float*)alloc((size_t)6291456 * 4);
  float* Hf0      = (float*)alloc((size_t)2097152 * 4);
  float* Hb0      = (float*)alloc((size_t)2097152 * 4);
  float* X1       = (float*)alloc((size_t)2097152 * 4);
  float* GxF1     = (float*)alloc((size_t)3145728 * 4);
  float* GxB1     = (float*)alloc((size_t)3145728 * 4);
  float* Hf1      = (float*)alloc((size_t)1048576 * 4);
  float* Hb1      = (float*)alloc((size_t)1048576 * 4);
  float* X2       = (float*)alloc((size_t)2097152 * 4);
  float* ghws     = (float*)alloc((size_t)2 * 32 * 768 * 4);
  float* meanb    = (float*)alloc(256 * 4);
  float* rstdb    = (float*)alloc(256 * 4);
  bf16_t* wihF0b  = (bf16_t*)alloc((size_t)1572864 * 2);
  bf16_t* wihB0b  = (bf16_t*)alloc((size_t)1572864 * 2);
  bf16_t* wihF1b  = (bf16_t*)alloc((size_t)393216 * 2);
  bf16_t* wihB1b  = (bf16_t*)alloc((size_t)393216 * 2);
  bf16_t* whhF0b  = (bf16_t*)alloc((size_t)196608 * 2);
  bf16_t* whhB0b  = (bf16_t*)alloc((size_t)196608 * 2);
  bf16_t* whhF1b  = (bf16_t*)alloc((size_t)196608 * 2);
  bf16_t* whhB1b  = (bf16_t*)alloc((size_t)196608 * 2);
  bf16_t* dcWihb  = (bf16_t*)alloc((size_t)393216 * 2);
  bf16_t* dcWhhb  = (bf16_t*)alloc((size_t)196608 * 2);
  bf16_t* att_bf  = (bf16_t*)alloc((size_t)32 * 512 * 2);
  float*  Gbuf    = (float*)alloc((size_t)32 * 768 * 4);
  float*  HNbuf   = (float*)alloc((size_t)32 * 256 * 4);
  float*  wbuf    = (float*)alloc((size_t)32 * 128 * 4);

  float* logits_out = (float*)d_out;           // (32,16,16)
  float* w_out      = (float*)d_out + 8192;    // (32,16,128)

  // ---- bf16 weight conversions ----
  k_f32_to_bf16<<<6144, 256, 0, stream>>>(f0_wih, wihF0b, 1572864);
  k_f32_to_bf16<<<6144, 256, 0, stream>>>(b0_wih, wihB0b, 1572864);
  k_f32_to_bf16<<<1536, 256, 0, stream>>>(f1_wih, wihF1b, 393216);
  k_f32_to_bf16<<<1536, 256, 0, stream>>>(b1_wih, wihB1b, 393216);
  k_f32_to_bf16<<<768, 256, 0, stream>>>(f0_whh, whhF0b, 196608);
  k_f32_to_bf16<<<768, 256, 0, stream>>>(b0_whh, whhB0b, 196608);
  k_f32_to_bf16<<<768, 256, 0, stream>>>(f1_whh, whhF1b, 196608);
  k_f32_to_bf16<<<768, 256, 0, stream>>>(b1_whh, whhB1b, 196608);
  k_f32_to_bf16<<<1536, 256, 0, stream>>>(dc_wih, dcWihb, 393216);
  k_f32_to_bf16<<<768, 256, 0, stream>>>(dc_whh, dcWhhb, 196608);

  // ---- conv stack ----
  k_conv_wmma<<<dim3(8192, 1), 256, 0, stream>>>(x, cw[0], cb[0], conv_buf,
                                                 256, 64, 1, 64);
  k_channel_stats<<<64, 256, 0, stream>>>(conv_buf, meanb, rstdb, 524288, 64);
  k_norm_pool<<<8192, 256, 0, stream>>>(conv_buf, meanb, rstdb, poolA,
                                        64, 32, (long)262144 * 64);
  k_conv_wmma<<<dim3(4096, 2), 256, 0, stream>>>(poolA, cw[1], cb[1], conv_buf,
                                                 256, 32, 64, 128);
  k_channel_stats<<<128, 256, 0, stream>>>(conv_buf, meanb, rstdb, 262144, 128);
  k_norm_pool<<<8192, 256, 0, stream>>>(conv_buf, meanb, rstdb, poolB,
                                        128, 16, (long)131072 * 128);
  k_conv_wmma<<<dim3(2048, 4), 256, 0, stream>>>(poolB, cw[2], cb[2], conv_buf,
                                                 256, 16, 128, 256);
  k_channel_stats<<<256, 256, 0, stream>>>(conv_buf, meanb, rstdb, 131072, 256);
  k_norm_pool<<<8192, 256, 0, stream>>>(conv_buf, meanb, rstdb, poolA,
                                        256, 8, (long)65536 * 256);
  k_transpose_x0<<<8192, 256, 0, stream>>>(poolA, X0, (long)8192 * 2048);

  // ---- GRU layer 0 ----
  k_gemm_wmma<<<dim3(128, 12), 256, 0, stream>>>(X0, wihF0b, f0_bih, GxF0,
                                                 8192, 768, 2048);
  k_gemm_wmma<<<dim3(128, 12), 256, 0, stream>>>(X0, wihB0b, b0_bih, GxB0,
                                                 8192, 768, 2048);
  k_gru_scan<<<2, 1024, 0, stream>>>(GxF0, GxB0, whhF0b, whhB0b, f0_bhh, b0_bhh,
                                     Hf0, Hb0, ghws, 256);
  k_concat_sub<<<4096, 256, 0, stream>>>(Hf0, Hb0, X1, 256, 128, 2,
                                         (long)32 * 128 * 512);

  // ---- GRU layer 1 ----
  k_gemm_wmma<<<dim3(64, 12), 256, 0, stream>>>(X1, wihF1b, f1_bih, GxF1,
                                                4096, 768, 512);
  k_gemm_wmma<<<dim3(64, 12), 256, 0, stream>>>(X1, wihB1b, b1_bih, GxB1,
                                                4096, 768, 512);
  k_gru_scan<<<2, 1024, 0, stream>>>(GxF1, GxB1, whhF1b, whhB1b, f1_bhh, b1_bhh,
                                     Hf1, Hb1, ghws, 128);
  k_concat_sub<<<4096, 256, 0, stream>>>(Hf1, Hb1, X2, 128, 128, 1,
                                         (long)32 * 128 * 512);

  // ---- attention decoder ----
  k_decoder<<<1, 1024, 0, stream>>>(X2, att_w, att_b, dcWihb, dc_bih, dcWhhb,
                                    dc_bhh, out_w, out_b, logits_out, w_out,
                                    att_bf, Gbuf, HNbuf, wbuf);
}
